// TitansMemoryLayer_39960375722539
// MI455X (gfx1250) — matrix operand order repaired and checked by reading, not compiled, not run on hardware
//
#include <hip/hip_runtime.h>
#include <hip/hip_bf16.h>

typedef __attribute__((ext_vector_type(16))) __bf16 v16bf;
typedef __attribute__((ext_vector_type(8)))  float  v8f;

#define B_  4
#define S_  4096
#define D_  1024
#define M_  4096
#define R_  (B_ * S_)          // 16384 rows (b*S + t)
#define NC_ 64                 // chunks per sequence
#define L_  64                 // chunk length (NC_*L_ == S_)
#define DECAY_F 0.9f
#define ALPHA_F 0.1f           // 1 - decay
#define DECAY_L 0.0011790184577738756f  // 0.9^64

#define PADK 40                // padded K-stride (halves) in LDS: 80B, multiple of 16B

static __device__ __forceinline__ unsigned short f2bf(float f) {
  union { float f; unsigned u; } v; v.f = f;
  unsigned r = v.u + 0x7FFFu + ((v.u >> 16) & 1u);  // round-to-nearest-even
  return (unsigned short)(r >> 16);
}

union FragU { uint4 q[2]; v16bf v; };

// ---------------- K0: weights fp32 -> bf16 ----------------
__global__ void cvt_weights(const float* __restrict__ w1, const float* __restrict__ w2,
                            unsigned short* __restrict__ w1b, unsigned short* __restrict__ w2b) {
  size_t i = (size_t)blockIdx.x * blockDim.x + threadIdx.x;
  const size_t n = (size_t)M_ * D_;
  if (i < n) w1b[i] = f2bf(w1[i]);
  else       w2b[i - n] = f2bf(w2[i - n]);
}

// ---------------- K1a: per-chunk local scan end values ----------------
__global__ void scan_chunk_ends(const float* __restrict__ x, float* __restrict__ carry) {
  int idx = blockIdx.x * blockDim.x + threadIdx.x;   // [0, B*NC*D)
  int d = idx & (D_ - 1);
  int k = (idx >> 10) & (NC_ - 1);
  int b = idx >> 16;
  const float* xp = x + ((size_t)b * S_ + (size_t)k * L_) * D_ + d;
  float c = 0.0f;
  #pragma unroll 4
  for (int i = 0; i < L_; ++i) c = DECAY_F * c + ALPHA_F * xp[(size_t)i * D_];
  carry[idx] = c;
}

// ---------------- K1b: serial scan over chunk carries (in place) ----------------
__global__ void scan_carries(float* __restrict__ carry) {
  int idx = blockIdx.x * blockDim.x + threadIdx.x;   // [0, B*D)
  int d = idx & (D_ - 1);
  int b = idx >> 10;
  float run = 0.0f;
  for (int k = 0; k < NC_; ++k) {
    size_t o = (((size_t)b * NC_ + k) << 10) + d;
    float e = carry[o];
    carry[o] = run;                 // chunk-start carry (state entering the chunk)
    run = DECAY_L * run + e;
  }
}

// ---------------- K1c: re-scan chunks with carry, emit prev (bf16) + final state ----------------
__global__ void scan_emit(const float* __restrict__ x, const float* __restrict__ carry,
                          unsigned short* __restrict__ prevb, float* __restrict__ finst) {
  int idx = blockIdx.x * blockDim.x + threadIdx.x;   // [0, B*NC*D)
  int d = idx & (D_ - 1);
  int k = (idx >> 10) & (NC_ - 1);
  int b = idx >> 16;
  const float* xp = x + ((size_t)b * S_ + (size_t)k * L_) * D_ + d;
  unsigned short* wr = prevb + ((size_t)b * S_ + (size_t)k * L_ + 1) * D_ + d;
  if (k == 0) prevb[(size_t)b * S_ * D_ + d] = 0;    // prev[b,0,:] = 0
  float c = carry[idx];
  #pragma unroll 4
  for (int i = 0; i < L_; ++i) {
    c = DECAY_F * c + ALPHA_F * xp[(size_t)i * D_];
    int t = k * L_ + i;
    if (t + 1 < S_) wr[(size_t)i * D_] = f2bf(c);    // prev[b,t+1,:] = c[b,t,:]
  }
  if (k == NC_ - 1) finst[b * D_ + d] = c;           // current_states[:, S-1, :]
}

// ---------------- shared GEMM core: block tile 128x256, BK=32, 8 waves x (64x64) ----
// A: (rows x K) bf16 row-major; Wt: (cols x K) bf16 row-major (operand B transposed).
// Global->LDS staging uses CDNA5 async DMA (global_load_async_to_lds_b128, ASYNCcnt),
// double-buffered; fragments come out of LDS with ds_load_b128.
template <int K>
static __device__ __forceinline__ void gemm_core(const unsigned short* __restrict__ A,
                                                 const unsigned short* __restrict__ Wt,
                                                 int rowBlock, int colBlock,
                                                 v8f (&acc)[4][4]) {
  __shared__ alignas(16) unsigned short sA[2][128 * PADK];
  __shared__ alignas(16) unsigned short sB[2][256 * PADK];

  const int tid  = threadIdx.x;
  const int lane = tid & 31;
  const int wave = tid >> 5;
  const int waveM = wave >> 2;          // 0..1
  const int waveN = wave & 3;           // 0..3
  const int lr = lane & 15;
  const int hi = lane >> 4;

  // Issue async global->LDS fills for one 32-K stage.
  auto issueA = [&](int st, int k0) {
    #pragma unroll
    for (int c = 0; c < 2; ++c) {
      int chunk = tid + c * 256;                   // 512 chunks of 8 halves
      int row = chunk >> 2, kc = (chunk & 3) * 8;
      const unsigned short* g = A + (size_t)(rowBlock + row) * K + k0 + kc;
      unsigned lds = (unsigned)(uintptr_t)&sA[st][row * PADK + kc];
      asm volatile("global_load_async_to_lds_b128 %0, %1, off"
                   :: "v"(lds), "v"(g) : "memory");
    }
  };
  auto issueB = [&](int st, int k0) {
    #pragma unroll
    for (int c = 0; c < 4; ++c) {
      int chunk = tid + c * 256;                   // 1024 chunks of 8 halves
      int col = chunk >> 2, kc = (chunk & 3) * 8;
      const unsigned short* g = Wt + (size_t)(colBlock + col) * K + k0 + kc;
      unsigned lds = (unsigned)(uintptr_t)&sB[st][col * PADK + kc];
      asm volatile("global_load_async_to_lds_b128 %0, %1, off"
                   :: "v"(lds), "v"(g) : "memory");
    }
  };

  issueA(0, 0);
  issueB(0, 0);

  for (int k0 = 0; k0 < K; k0 += 32) {
    const int st = (k0 >> 5) & 1;
    asm volatile("s_wait_asynccnt 0x0" ::: "memory");  // my fills for stage st done
    __syncthreads();                                   // everyone's fills visible; dscnt drained
    if (k0 + 32 < K) {                                 // prefetch next stage into other buffer
      issueA(st ^ 1, k0 + 32);
      issueB(st ^ 1, k0 + 32);
    }

    FragU a[4], bf[4];
    #pragma unroll
    for (int i = 0; i < 4; ++i) {
      const unsigned short* p = &sA[st][(waveM * 64 + i * 16 + lr) * PADK + hi * 8];
      a[i].q[0] = *reinterpret_cast<const uint4*>(p);        // K = base+0..7
      a[i].q[1] = *reinterpret_cast<const uint4*>(p + 16);   // K = base+16..23
    }
    #pragma unroll
    for (int j = 0; j < 4; ++j) {
      const unsigned short* p = &sB[st][(waveN * 64 + j * 16 + lr) * PADK + hi * 8];
      bf[j].q[0] = *reinterpret_cast<const uint4*>(p);
      bf[j].q[1] = *reinterpret_cast<const uint4*>(p + 16);
    }
    #pragma unroll
    for (int i = 0; i < 4; ++i)
      #pragma unroll
      for (int j = 0; j < 4; ++j)
        acc[i][j] = __builtin_amdgcn_wmma_f32_16x16x32_bf16(
            false, a[i].v, false, bf[j].v, (short)0, acc[i][j], false, false);
  }
}

// ---------------- K2: H = silu(prev @ w1^T) ----------------
__global__ void __launch_bounds__(256)
gemm1_silu(const unsigned short* __restrict__ A,     // prev (R_ x D_) bf16
           const unsigned short* __restrict__ W,     // w1   (M_ x D_) bf16
           unsigned short* __restrict__ H) {         // out  (R_ x M_) bf16
  const int nbN = M_ / 256;
  const int rowBlock = (blockIdx.x / nbN) * 128;
  const int colBlock = (blockIdx.x % nbN) * 256;
  const int lane = threadIdx.x & 31;
  const int wave = threadIdx.x >> 5;
  const int lr = lane & 15, hi = lane >> 4;

  const v8f zero = {0.f,0.f,0.f,0.f,0.f,0.f,0.f,0.f};
  v8f acc[4][4];
  #pragma unroll
  for (int i = 0; i < 4; ++i)
    #pragma unroll
    for (int j = 0; j < 4; ++j) acc[i][j] = zero;

  gemm_core<D_>(A, W, rowBlock, colBlock, acc);

  const int rowBase = rowBlock + (wave >> 2) * 64;
  const int colBase = colBlock + (wave & 3) * 64;
  #pragma unroll
  for (int i = 0; i < 4; ++i)
    #pragma unroll
    for (int j = 0; j < 4; ++j) {
      const int col = colBase + j * 16 + lr;
      #pragma unroll
      for (int e = 0; e < 8; ++e) {
        const int row = rowBase + i * 16 + hi * 8 + e;
        float v = acc[i][j][e];
        float s = v * (1.0f / (1.0f + __expf(-v)));          // silu
        H[(size_t)row * M_ + col] = f2bf(s);
      }
    }
}

// ---------------- K3: out = H @ w2^T (+ fused squared-error partials) ----------------
__global__ void __launch_bounds__(256)
gemm2_out(const unsigned short* __restrict__ H,      // (R_ x M_) bf16
          const unsigned short* __restrict__ W2,     // (D_ x M_) bf16
          const float* __restrict__ x,               // (R_ x D_) fp32
          float* __restrict__ out,                   // (R_ x D_) fp32
          float* __restrict__ partials) {            // one per wave (4096)
  const int nbN = D_ / 256;
  const int rowBlock = (blockIdx.x / nbN) * 128;
  const int colBlock = (blockIdx.x % nbN) * 256;
  const int lane = threadIdx.x & 31;
  const int wave = threadIdx.x >> 5;
  const int lr = lane & 15, hi = lane >> 4;

  const v8f zero = {0.f,0.f,0.f,0.f,0.f,0.f,0.f,0.f};
  v8f acc[4][4];
  #pragma unroll
  for (int i = 0; i < 4; ++i)
    #pragma unroll
    for (int j = 0; j < 4; ++j) acc[i][j] = zero;

  gemm_core<M_>(H, W2, rowBlock, colBlock, acc);

  const int rowBase = rowBlock + (wave >> 2) * 64;
  const int colBase = colBlock + (wave & 3) * 64;
  float part = 0.0f;
  #pragma unroll
  for (int i = 0; i < 4; ++i)
    #pragma unroll
    for (int j = 0; j < 4; ++j) {
      const int col = colBase + j * 16 + lr;
      #pragma unroll
      for (int e = 0; e < 8; ++e) {
        const int row = rowBase + i * 16 + hi * 8 + e;
        float v = acc[i][j][e];
        out[(size_t)row * D_ + col] = v;
        float dvd = v - x[(size_t)row * D_ + col];
        part += dvd * dvd;
      }
    }
  #pragma unroll
  for (int off = 16; off > 0; off >>= 1) part += __shfl_xor(part, off, 32);
  if (lane == 0) partials[blockIdx.x * 8 + wave] = part;
}

// ---------------- K4: deterministic loss reduction ----------------
__global__ void reduce_loss(const float* __restrict__ partials, float* __restrict__ loss) {
  __shared__ float sm[256];
  float s = 0.0f;
  for (int i = threadIdx.x; i < 4096; i += 256) s += partials[i];
  sm[threadIdx.x] = s;
  __syncthreads();
  for (int off = 128; off > 0; off >>= 1) {
    if ((int)threadIdx.x < off) sm[threadIdx.x] += sm[threadIdx.x + off];
    __syncthreads();
  }
  if (threadIdx.x == 0) *loss = sm[0] * (1.0f / ((float)R_ * (float)D_));
}

extern "C" void kernel_launch(void* const* d_in, const int* in_sizes, int n_in,
                              void* d_out, int out_size, void* d_ws, size_t ws_size,
                              hipStream_t stream) {
  const float* x  = (const float*)d_in[0];
  const float* w1 = (const float*)d_in[1];
  const float* w2 = (const float*)d_in[2];

  float* out   = (float*)d_out;                 // mem_out: R_*D_ floats
  float* loss  = out + (size_t)R_ * D_;         // scalar
  float* finst = loss + 1;                      // final_state: B_*D_ floats

  char* p = (char*)d_ws;
  unsigned short* w1b   = (unsigned short*)p;  p += (size_t)M_ * D_ * 2;   // 8 MB
  unsigned short* w2b   = (unsigned short*)p;  p += (size_t)D_ * M_ * 2;   // 8 MB
  unsigned short* prevb = (unsigned short*)p;  p += (size_t)R_ * D_ * 2;   // 32 MB
  unsigned short* hb    = (unsigned short*)p;  p += (size_t)R_ * M_ * 2;   // 128 MB
  float* carry          = (float*)p;           p += (size_t)B_ * NC_ * D_ * 4; // 1 MB
  float* partials       = (float*)p;                                       // 16 KB

  cvt_weights    <<<(2u * M_ * D_) / 256, 256, 0, stream>>>(w1, w2, w1b, w2b);
  scan_chunk_ends<<<(B_ * NC_ * D_) / 256, 256, 0, stream>>>(x, carry);
  scan_carries   <<<(B_ * D_) / 256,       256, 0, stream>>>(carry);
  scan_emit      <<<(B_ * NC_ * D_) / 256, 256, 0, stream>>>(x, carry, prevb, finst);
  gemm1_silu     <<<(R_ / 128) * (M_ / 256), 256, 0, stream>>>(prevb, w1b, hb);
  gemm2_out      <<<(R_ / 128) * (D_ / 256), 256, 0, stream>>>(hb, w2b, x, out, partials);
  reduce_loss    <<<1, 256, 0, stream>>>(partials, loss);
}
